// MGNConv_72894184948202
// MI455X (gfx1250) — matrix operand order repaired and checked by reading, not compiled, not run on hardware
//
#include <hip/hip_runtime.h>
#include <hip/hip_bf16.h>
#include <math.h>

// ---------------------------------------------------------------------------
// MGNConv (scalar irreps) for MI455X / gfx1250.
// Dominant cost: per-edge radial GEMM [E,64]@[64,512] fused with the scalar
// tensor-product contraction -> done with v_wmma_f32_16x16x32_bf16, one wave
// per 16-edge block, w2 staged in LDS (bf16, transposed, norms folded).
// ---------------------------------------------------------------------------

typedef __attribute__((ext_vector_type(16))) __bf16 v16bf;
typedef __attribute__((ext_vector_type(8)))  float  v8f;
typedef __attribute__((ext_vector_type(4)))  float  f32x4;

union V16BF { v16bf v; f32x4 q[2]; __bf16 h[16]; };

#define WPB 8                       // waves per 256-thread workgroup
#define W2T_ELEMS (512 * 64)        // bf16, layout [col(512)][c(64)]
#define W1T_ELEMS (64 * 16)         // bf16, layout [col(64)][c(16)]
// per-wave scratch: h [16 edges][64] bf16, uT [64 m][16 edges] f32, j [16] int
#define PW_BYTES (16 * 64 * 2 + 64 * 16 * 4 + 16 * 4)

// fast silu: x * sigmoid(x) with native v_exp_f32 (base-2) + v_rcp_f32
__device__ __forceinline__ float fast_silu(float x) {
  const float e = __builtin_amdgcn_exp2f(-x * 1.4426950408889634f);
  return x * __builtin_amdgcn_rcpf(1.0f + e);
}

__global__ void __launch_bounds__(256) zero_f32_kernel(float* p, int n) {
  int i = blockIdx.x * blockDim.x + threadIdx.x;
  if (i < n) p[i] = 0.0f;
}

__global__ void __launch_bounds__(256) edge_kernel(
    const float* __restrict__ x,      // [N,8]
    const int*   __restrict__ eidx,   // [2,E]
    const float* __restrict__ attr,   // [E,4]
    const float* __restrict__ emb,    // [E,16]
    const float* __restrict__ w1,     // [16,64]
    const float* __restrict__ w2,     // [64,512]
    float* __restrict__ edge_out,     // [E,8]
    float* __restrict__ esum,         // [N,8]  (pre-zeroed)
    int N, int E)
{
  extern __shared__ char smem[];
  __bf16* w2t = (__bf16*)smem;                          // 64 KB
  __bf16* w1t = w2t + W2T_ELEMS;                        // 2 KB
  char* pwbase = (char*)(w1t + W1T_ELEMS);

  const int tid = threadIdx.x;

  // ---- stage weights into LDS as bf16 with path norms folded ---------------
  // w2t[col][c] = w2[c][col] / (sqrt(64) * sqrt(4*16)) = w2[c][col] / 64
  for (int t = tid; t < 512 * 64; t += blockDim.x) {
    int c = t >> 9, col = t & 511;
    w2t[col * 64 + c] = (__bf16)(w2[t] * (1.0f / 64.0f));
  }
  // w1t[col][c] = w1[c][col] / sqrt(16)
  for (int t = tid; t < 16 * 64; t += blockDim.x) {
    int c = t >> 6, col = t & 63;
    w1t[col * 16 + c] = (__bf16)(w1[t] * 0.25f);
  }
  __syncthreads();

  const int wave  = tid >> 5;
  const int lane  = tid & 31;
  const int laneM = lane & 15;
  const int hi    = lane >> 4;      // 0: rows/K-low half, 1: rows/K-high half

  __bf16* hB = (__bf16*)(pwbase + wave * PW_BYTES);     // [16 edges][64] bf16
  float*  uT = (float*)(hB + 16 * 64);                  // [64 m][16 edges] f32
  int*    jB = (int*)(uT + 64 * 16);                    // [16]

  // ---- hoist GEMM1 B tiles (w1t is loop-invariant): 4 tiles x 8 VGPRs ------
  // B layout (16-bit 32x16): elems 0-7 = K (hi?8..15:0..7) (live),
  // elems 8-15 = K 16..31 half -> zero pad.
  V16BF bw1[4];
  #pragma unroll
  for (int t1 = 0; t1 < 4; ++t1) {
    {
      const f32x4* wp = (const f32x4*)(w1t + (t1 * 16 + laneM) * 16 + hi * 8);
      bw1[t1].q[0] = wp[0];
      bw1[t1].q[1] = f32x4{0.f, 0.f, 0.f, 0.f};
    }
  }

  const int nblk    = E >> 4;                 // E is a multiple of 16
  const int gw0     = blockIdx.x * WPB + wave;
  const int gstride = gridDim.x * WPB;

  for (int blk = gw0; blk < nblk; blk += gstride) {
    const int e0 = blk << 4;

    // ---- step 1: uT[a*16+b][e] = attr[e][a] * z[e][b], z = cat(x[i],x[j]) --
    {
      const int row  = lane >> 1;     // edge within block (0..15)
      const int half = lane & 1;      // 0 -> x[i] (b 0..7), 1 -> x[j] (b 8..15)
      const int e    = e0 + row;
      const int src  = half ? eidx[E + e] : eidx[e];
      if (half) jB[row] = src;
      const f32x4 a4 = *(const f32x4*)(attr + (size_t)e * 4);
      const f32x4 z0 = *(const f32x4*)(x + (size_t)src * 8);
      const f32x4 z1 = *(const f32x4*)(x + (size_t)src * 8 + 4);
      #pragma unroll
      for (int a = 0; a < 4; ++a) {
        const float s = a4[a];
        float* uc = uT + (a * 16 + half * 8) * 16 + row;
        uc[0 * 16] = s * z0.x;  uc[1 * 16] = s * z0.y;
        uc[2 * 16] = s * z0.z;  uc[3 * 16] = s * z0.w;
        uc[4 * 16] = s * z1.x;  uc[5 * 16] = s * z1.y;
        uc[6 * 16] = s * z1.z;  uc[7 * 16] = s * z1.w;
      }
    }

    // ---- step 2: h = silu(emb @ (w1/4)), K=16 zero-padded to 32 ------------
    // A layout (16-bit 16x32): lane holds M=laneM; elems 0-7 = K (hi?8..15:0..7),
    // elems 8-15 = K 16..31 half -> zero pad.
    V16BF ag;
    {
      const float* er = emb + (size_t)(e0 + laneM) * 16 + hi * 8;
      const f32x4 ev0 = *(const f32x4*)er;
      const f32x4 ev1 = *(const f32x4*)(er + 4);
      #pragma unroll
      for (int t = 0; t < 4; ++t) ag.h[t]     = (__bf16)ev0[t];
      #pragma unroll
      for (int t = 0; t < 4; ++t) ag.h[4 + t] = (__bf16)ev1[t];
      #pragma unroll
      for (int t = 8; t < 16; ++t) ag.h[t] = (__bf16)0.0f;
    }
    #pragma unroll
    for (int t1 = 0; t1 < 4; ++t1) {
      const int col = t1 * 16 + laneM;
      v8f c = {0.f, 0.f, 0.f, 0.f, 0.f, 0.f, 0.f, 0.f};
      c = __builtin_amdgcn_wmma_f32_16x16x32_bf16(false, ag.v, false, bw1[t1].v,
                                                  (short)0, c, false, false);
      #pragma unroll
      for (int v = 0; v < 8; ++v) {
        hB[(v + hi * 8) * 64 + col] = (__bf16)fast_silu(c[v]);
      }
    }

    // ---- step 3: t = h @ (w2/64), fused with  edge_out += u * t ------------
    V16BF A0, A1;
    {
      const __bf16* hr = hB + laneM * 64;
      const int b0 = hi * 8;            // K-step 0: runs {b0..b0+7, b0+16..}
      A0.q[0] = *(const f32x4*)(hr + b0);
      A0.q[1] = *(const f32x4*)(hr + b0 + 16);
      const int b1 = 32 + hi * 8;       // K-step 1
      A1.q[0] = *(const f32x4*)(hr + b1);
      A1.q[1] = *(const f32x4*)(hr + b1 + 16);
    }
    v8f p = {0.f, 0.f, 0.f, 0.f, 0.f, 0.f, 0.f, 0.f};
    const int mpar = laneM >> 3;        // this lane's m-parity within a tile
    #pragma unroll 4
    for (int nt = 0; nt < 32; ++nt) {
      // B layout (16-bit 32x16): lane col = nt*16+laneM; runs
      // {hi*8..hi*8+7, hi*8+16..hi*8+23} in w2t[col][...]
      const __bf16* wcol = w2t + (size_t)(nt * 16 + laneM) * 64;
      V16BF B0, B1;
      B0.q[0] = *(const f32x4*)(wcol + hi * 8);
      B0.q[1] = *(const f32x4*)(wcol + hi * 8 + 16);
      B1.q[0] = *(const f32x4*)(wcol + 32 + hi * 8);
      B1.q[1] = *(const f32x4*)(wcol + 32 + hi * 8 + 16);
      v8f acc = {0.f, 0.f, 0.f, 0.f, 0.f, 0.f, 0.f, 0.f};
      acc = __builtin_amdgcn_wmma_f32_16x16x32_bf16(false, A0.v, false, B0.v,
                                                    (short)0, acc, false, false);
      acc = __builtin_amdgcn_wmma_f32_16x16x32_bf16(false, A1.v, false, B1.v,
                                                    (short)0, acc, false, false);
      // column = nt*16 + laneM -> m = 2*nt + mpar, k = lane&7 (fixed per lane)
      const float* up = uT + (size_t)(nt * 2 + mpar) * 16 + hi * 8;
      const f32x4 ua = *(const f32x4*)up;
      const f32x4 ub = *(const f32x4*)(up + 4);
      p[0] += ua.x * acc[0];  p[1] += ua.y * acc[1];
      p[2] += ua.z * acc[2];  p[3] += ua.w * acc[3];
      p[4] += ub.x * acc[4];  p[5] += ub.y * acc[5];
      p[6] += ub.z * acc[6];  p[7] += ub.w * acc[7];
    }

    // ---- combine the two m-parities, store + scatter -----------------------
    #pragma unroll
    for (int v = 0; v < 8; ++v) p[v] += __shfl_xor(p[v], 8);
    if (laneM < 8) {                    // lanes {0..7, 16..23} hold full sums
      const int k = lane & 7;
      #pragma unroll
      for (int v = 0; v < 8; ++v) {
        const int row = v + hi * 8;     // edge row (lanes>=16 cover rows 8-15)
        const float val = p[v];
        edge_out[(size_t)(e0 + row) * 8 + k] = val;
        atomicAdd(esum + (size_t)jB[row] * 8 + k, val);   // no-return f32 add
      }
    }
  }
}

// node_tp: x_out[n,k] = (1/8) * sum_{a,b} x[n,a] * esum[n,b] * wn[a,b,k]
__global__ void __launch_bounds__(256) node_kernel(
    const float* __restrict__ x, const float* __restrict__ esum,
    const float* __restrict__ wn, float* __restrict__ xout, int N)
{
  __shared__ float wns[8 * 8 * 16];
  for (int t = threadIdx.x; t < 8 * 8 * 16; t += blockDim.x) wns[t] = wn[t];
  __syncthreads();
  const int n = blockIdx.x * blockDim.x + threadIdx.x;
  if (n >= N) return;
  float xv[8], ev[8], pm[64];
  #pragma unroll
  for (int a = 0; a < 8; ++a) xv[a] = x[(size_t)n * 8 + a];
  #pragma unroll
  for (int b = 0; b < 8; ++b) ev[b] = esum[(size_t)n * 8 + b];
  #pragma unroll
  for (int a = 0; a < 8; ++a)
    #pragma unroll
    for (int b = 0; b < 8; ++b) pm[a * 8 + b] = xv[a] * ev[b];
  #pragma unroll
  for (int k = 0; k < 16; ++k) {
    float s = 0.0f;
    #pragma unroll
    for (int m = 0; m < 64; ++m) s += pm[m] * wns[m * 16 + k];
    xout[(size_t)n * 16 + k] = s * 0.125f;
  }
}

extern "C" void kernel_launch(void* const* d_in, const int* in_sizes, int n_in,
                              void* d_out, int out_size, void* d_ws, size_t ws_size,
                              hipStream_t stream) {
  const float* x    = (const float*)d_in[0];
  const int*   eidx = (const int*)  d_in[1];
  const float* attr = (const float*)d_in[2];
  const float* emb  = (const float*)d_in[3];
  const float* w1   = (const float*)d_in[4];
  const float* w2   = (const float*)d_in[5];
  const float* wn   = (const float*)d_in[6];
  const int N = in_sizes[0] / 8;
  const int E = in_sizes[2] / 4;

  float* dout = (float*)d_out;
  float* xout = dout;                        // [N,16]
  float* eout = dout + (size_t)N * 16;       // [E,8]
  float* esum = (float*)d_ws;                // [N,8] scratch

  zero_f32_kernel<<<(N * 8 + 255) / 256, 256, 0, stream>>>(esum, N * 8);

  const size_t smem = (size_t)W2T_ELEMS * 2 + (size_t)W1T_ELEMS * 2 +
                      (size_t)WPB * PW_BYTES;              // ~117 KB
  const int nblk = E >> 4;
  int grid = (nblk + WPB - 1) / WPB;
  if (grid > 1024) grid = 1024;              // persistent, grid-stride per wave
  if (grid < 1) grid = 1;
  edge_kernel<<<grid, 256, smem, stream>>>(x, eidx, attr, emb, w1, w2,
                                           eout, esum, N, E);

  node_kernel<<<(N + 255) / 256, 256, 0, stream>>>(x, esum, wn, xout, N);
}
